// Attention_53437983096950
// MI455X (gfx1250) — matrix operand order repaired and checked by reading, not compiled, not run on hardware
//
#include <hip/hip_runtime.h>
#include <cstdint>

// ---------------------------------------------------------------------------
// CDNA5 (gfx1250) wave32 WMMA bf16 implementation of sliding-window GQA block,
// with Tensor Data Mover (TDM) global->LDS tile DMA in the attention kernel.
// ---------------------------------------------------------------------------

typedef __attribute__((ext_vector_type(16))) __bf16    v16bf;
typedef __attribute__((ext_vector_type(8)))  float     v8f;
typedef __attribute__((ext_vector_type(4)))  uint32_t  v4u;
typedef __attribute__((ext_vector_type(8)))  uint32_t  v8u;

union FragB { uint32_t u[8]; v16bf v; };

__device__ __forceinline__ uint32_t pack2bf(float a, float b) {
  // round-to-nearest-even f32 -> bf16, packed pair
  uint32_t ua = __float_as_uint(a), ub = __float_as_uint(b);
  ua += 0x7FFFu + ((ua >> 16) & 1u);
  ub += 0x7FFFu + ((ub >> 16) & 1u);
  return (ua >> 16) | (ub & 0xFFFF0000u);
}

__device__ __forceinline__ unsigned short f2bf(float a) {
  uint32_t u = __float_as_uint(a);
  u += 0x7FFFu + ((u >> 16) & 1u);
  return (unsigned short)(u >> 16);
}

// A-matrix 16x32 bf16 fragment (ISA 7.12.2): lane half h, dword r holds K pair
// starting at bf16 offset 8h + (r<4 ? 2r : 2r+8)  -> dword offsets {0..3,8..11}+4h
// p points at LDS dword row base (row = M = lane%16), K-contiguous.
__device__ __forceinline__ v16bf ldsA(const uint32_t* p, int lh) {
  FragB f; p += 4 * lh;
  f.u[0] = p[0]; f.u[1] = p[1]; f.u[2] = p[2];  f.u[3] = p[3];
  f.u[4] = p[8]; f.u[5] = p[9]; f.u[6] = p[10]; f.u[7] = p[11];
  return f.v;
}

// B-matrix 32x16 bf16 fragment: lane holds column N = lane%16; dword r holds
// K pair at bf16 offset 8h + (r<4 ? 2r : 2r+8) -> dwords {0..3,8..11}+4h. p points at LDS row base of
// that column stored K-contiguous ("B stored transposed": [N][K]).
__device__ __forceinline__ v16bf ldsB(const uint32_t* p, int lh) {
  FragB f; p += 4 * lh;
#pragma unroll
  for (int r = 0; r < 8; ++r) f.u[r] = p[r < 4 ? r : r + 4];
  return f.v;
}

__device__ __forceinline__ v8f wmma_bf16(v16bf a, v16bf b, v8f c) {
  return __builtin_amdgcn_wmma_f32_16x16x32_bf16(false, a, false, b,
                                                 (short)0, c, false, false);
}

// ---------------------------------------------------------------------------
// TDM: DMA a 2D tile of bf16 (tile_w elems x tile_h rows) from global memory
// (row stride `stride_elems`) into LDS at byte offset lds_off, with LDS
// padding of 4 DWORDs after every 64 DWORDs (=> LDS row stride 68 dwords,
// matching the fragment-gather layout used below). Tile must be in-bounds
// (tensor dims are set equal to the tile so TDM's OOB logic never triggers).
// Descriptor per CDNA5 ISA (D# groups 0/1, 2D, no cluster).
// Wave-level: caller gates issue to one wave; completion via TENSORcnt.
// ---------------------------------------------------------------------------
__device__ __forceinline__ void tdm_load_2d_bf16(uint32_t lds_off,
                                                 const void* gsrc,
                                                 uint32_t tile_w, uint32_t tile_h,
                                                 uint32_t stride_elems)
{
  const uint64_t ga = (uint64_t)(uintptr_t)gsrc;
  v4u g0;
  g0[0] = 1u;                                        // count=1 (valid), user mode
  g0[1] = lds_off;                                   // lds_addr (bytes)
  g0[2] = (uint32_t)(ga & 0xFFFFFFFFu);              // global_addr[31:0]
  g0[3] = (uint32_t)((ga >> 32) & 0x01FFFFFFu)       // global_addr[56:32]
        | (2u << 30);                                // type = 2 ("image")
  v8u g1;
  g1[0] = (1u << 16)                                 // data_size = 1 (2 bytes)
        | (1u << 20)                                 // pad_enable
        | (5u << 22)                                 // pad_interval: 64 DWORDs
        | (3u << 25);                                // pad_amount: 4 DWORDs
  g1[1] = (tile_w & 0xFFFFu) << 16;                  // tensor_dim0[15:0]
  g1[2] = ((tile_w >> 16) & 0xFFFFu)                 // tensor_dim0[31:16]
        | ((tile_h & 0xFFFFu) << 16);                // tensor_dim1[15:0]
  g1[3] = ((tile_h >> 16) & 0xFFFFu)                 // tensor_dim1[31:16]
        | (tile_w << 16);                            // tile_dim0
  g1[4] = tile_h;                                    // tile_dim1 (tile_dim2 = 0)
  g1[5] = stride_elems;                              // tensor_dim0_stride[31:0]
  g1[6] = 0;                                         // stride[47:32], dim1_stride lo
  g1[7] = 0;
  asm volatile("tensor_load_to_lds %0, %1" :: "s"(g0), "s"(g1) : "memory");
}

// ---------------------------------------------------------------------------
// C[M,N] = A[M,K] @ B[N,K]^T   (both row-major f32, bf16 WMMA compute)
// Block tile 128x64, BK=32. 256 threads = 8 waves, each wave 32x32.
// M%128==0, N%64==0, K%32==0 assumed (true for all calls here).
// ---------------------------------------------------------------------------
__global__ __launch_bounds__(256) void gemm_xwT_bf16(
    const float* __restrict__ A, const float* __restrict__ B,
    float* __restrict__ C, int M, int N, int K)
{
  __shared__ uint32_t As[128 * 20];  // 128 rows x 32 bf16 (16 dw) + pad -> stride 20 dw
  __shared__ uint32_t Bs[64 * 20];

  const int tid  = threadIdx.x;
  const int lane = tid & 31, w = tid >> 5;
  const int lh = lane >> 4, ll = lane & 15;
  const int m0 = blockIdx.y * 128, n0 = blockIdx.x * 64;
  const int wm = (w >> 1) * 32, wn = (w & 1) * 32;

  v8f acc[2][2];
#pragma unroll
  for (int i = 0; i < 2; ++i)
#pragma unroll
    for (int j = 0; j < 2; ++j) acc[i][j] = (v8f){0,0,0,0,0,0,0,0};

  const int arow = tid >> 1, ahalf = tid & 1;   // A: 2 threads/row, 16 f32 each
  const int brow = tid >> 2, bq    = tid & 3;   // B: 4 threads/row, 8 f32 each

  for (int k0 = 0; k0 < K; k0 += 32) {
    __syncthreads();
    {
      const float4* s4 = (const float4*)(A + (size_t)(m0 + arow) * K + k0 + ahalf * 16);
      __builtin_prefetch(s4 + 8, 0, 0);  // speculative next-tile prefetch
      float4 a0 = s4[0], a1 = s4[1], a2 = s4[2], a3 = s4[3];
      uint4* d = (uint4*)(As + arow * 20 + ahalf * 8);
      d[0] = make_uint4(pack2bf(a0.x, a0.y), pack2bf(a0.z, a0.w),
                        pack2bf(a1.x, a1.y), pack2bf(a1.z, a1.w));
      d[1] = make_uint4(pack2bf(a2.x, a2.y), pack2bf(a2.z, a2.w),
                        pack2bf(a3.x, a3.y), pack2bf(a3.z, a3.w));
    }
    {
      const float4* s4 = (const float4*)(B + (size_t)(n0 + brow) * K + k0 + bq * 8);
      float4 b0 = s4[0], b1 = s4[1];
      uint4* d = (uint4*)(Bs + brow * 20 + bq * 4);
      d[0] = make_uint4(pack2bf(b0.x, b0.y), pack2bf(b0.z, b0.w),
                        pack2bf(b1.x, b1.y), pack2bf(b1.z, b1.w));
    }
    __syncthreads();

    v16bf a0 = ldsA(As + (wm + ll) * 20, lh);
    v16bf a1 = ldsA(As + (wm + 16 + ll) * 20, lh);
    v16bf b0 = ldsB(Bs + (wn + ll) * 20, lh);
    v16bf b1 = ldsB(Bs + (wn + 16 + ll) * 20, lh);
    acc[0][0] = wmma_bf16(a0, b0, acc[0][0]);
    acc[0][1] = wmma_bf16(a0, b1, acc[0][1]);
    acc[1][0] = wmma_bf16(a1, b0, acc[1][0]);
    acc[1][1] = wmma_bf16(a1, b1, acc[1][1]);
  }

  // C/D layout: vgpr r -> row r + 8*(lane/16), col = lane%16
#pragma unroll
  for (int rt = 0; rt < 2; ++rt)
#pragma unroll
    for (int ct = 0; ct < 2; ++ct) {
      float* cp = C + (size_t)(m0 + wm + rt * 16 + 8 * lh) * N + n0 + wn + ct * 16 + ll;
#pragma unroll
      for (int r = 0; r < 8; ++r) cp[(size_t)r * N] = acc[rt][ct][r];
    }
}

// ---------------------------------------------------------------------------
// rmsnorm + rope over DH=128 per (t, head); one wave per row.
// rows: t in [0,4096), hh in [0,20): hh<16 -> q head, else k head hh-16.
// Writes bf16 (packed pairs) to qb/kb.
// ---------------------------------------------------------------------------
__global__ __launch_bounds__(256) void rmsnorm_rope_kernel(
    const float* __restrict__ qf, const float* __restrict__ kf,
    uint32_t* __restrict__ qb, uint32_t* __restrict__ kb)
{
  const int row  = blockIdx.x * 8 + (threadIdx.x >> 5);
  const int lane = threadIdx.x & 31;
  const int t = row / 20, hh = row % 20;
  const float* src; uint32_t* dst;
  if (hh < 16) { src = qf + ((size_t)t * 16 + hh) * 128;        dst = qb + ((size_t)t * 16 + hh) * 64; }
  else         { src = kf + ((size_t)t * 4 + (hh - 16)) * 128;  dst = kb + ((size_t)t * 4 + (hh - 16)) * 64; }

  float4 x = ((const float4*)src)[lane];
  float ss = x.x * x.x + x.y * x.y + x.z * x.z + x.w * x.w;
  ss += __shfl_xor(ss, 16); ss += __shfl_xor(ss, 8);
  ss += __shfl_xor(ss, 4);  ss += __shfl_xor(ss, 2);
  ss += __shfl_xor(ss, 1);
  const float rn = rsqrtf(ss * (1.0f / 128.0f) + 1e-6f);

  const float x1a = x.x * rn, x2a = x.y * rn;   // pair d0 = 2*lane
  const float x1b = x.z * rn, x2b = x.w * rn;   // pair d1 = 2*lane+1
  const float NEG_LN_TH_64 = -0.14391156605f;   // -ln(10000)/64
  const int d0 = 2 * lane, d1 = 2 * lane + 1;
  const float ang0 = (float)t * __expf(NEG_LN_TH_64 * (float)d0);
  const float ang1 = (float)t * __expf(NEG_LN_TH_64 * (float)d1);
  float s0, c0, s1, c1;
  sincosf(ang0, &s0, &c0);
  sincosf(ang1, &s1, &c1);

  const float o1a = x1a * c0 - x2a * s0, o2a = x1a * s0 + x2a * c0;
  const float o1b = x1b * c1 - x2b * s1, o2b = x1b * s1 + x2b * c1;
  dst[lane]      = pack2bf(o1a, o1b);   // out[d0], out[d1]
  dst[32 + lane] = pack2bf(o2a, o2b);   // out[64+d0], out[64+d1]
}

// ---------------------------------------------------------------------------
// elementwise f32 -> packed bf16 (for V)
// ---------------------------------------------------------------------------
__global__ __launch_bounds__(256) void cvt_f32_bf16x2(
    const float* __restrict__ in, uint32_t* __restrict__ out, int n)
{
  int i = blockIdx.x * 256 + threadIdx.x;
  if (i < n) { float2 f = ((const float2*)in)[i]; out[i] = pack2bf(f.x, f.y); }
}

// ---------------------------------------------------------------------------
// Sliding-window attention, flash-style online softmax.
// Block = (head h, chunk c, q-subtile of 128 rows). 8 waves x 16 q rows each.
// Q and K tiles arrive via TDM (tensor_load_to_lds) with LDS row padding;
// S = Q@K^T and O += P@V both via bf16 WMMA.
// ---------------------------------------------------------------------------
__global__ __launch_bounds__(256) void attn_kernel(
    const uint32_t* __restrict__ qb, const uint32_t* __restrict__ kb,
    const uint32_t* __restrict__ vb, float* __restrict__ y)
{
  __shared__ __align__(16) uint32_t Qs[128 * 68];      // 128 x 128 bf16 + 4dw pad/row
  __shared__ __align__(16) uint32_t Ks[32 * 68];       // 32 keys x 128 bf16 + pad
  __shared__ __align__(16) uint32_t Vt[128 * 20];      // transposed: [dh][32 keys + pad]
  __shared__ __align__(16) uint32_t Pls[8 * 16 * 18];  // per-wave P: 16 x 32 bf16 + pad

  const int b = blockIdx.x;
  const int h = b >> 5, c = (b >> 2) & 7, sub = b & 3;
  const int qs  = c * 512 + sub * 128;
  const int kvh = h >> 2;                // GQA: head h -> kv head h/4
  const int tid = threadIdx.x, w = tid >> 5, lane = tid & 31;
  const int lh = lane >> 4, ll = lane & 15;

  // Q tile (rows qs..qs+127, head h) via TDM: 128x128 bf16, global stride 2048
  if (w == 0) {
    tdm_load_2d_bf16((uint32_t)(uintptr_t)Qs,
                     (const unsigned short*)qb + (size_t)qs * 2048 + h * 128,
                     128, 128, 2048);
  }

  v8f o[8];
#pragma unroll
  for (int i = 0; i < 8; ++i) o[i] = (v8f){0,0,0,0,0,0,0,0};
  float m8[8], l8[8];
#pragma unroll
  for (int r = 0; r < 8; ++r) { m8[r] = -3.0e38f; l8[r] = 0.0f; }

  const int lo   = max(0, c * 512 - 1023);    // window lower bound (per chunk)
  const int klo0 = lo & ~31;
  const float SCALE = 0.08838834764831845f;   // 1/sqrt(128)
  unsigned short* PlsS = (unsigned short*)Pls;
  unsigned short* VtS  = (unsigned short*)Vt;

  for (int kstart = klo0; kstart < qs + 128; kstart += 32) {
    __syncthreads();  // previous iteration done with Ks/Vt
    // K block via TDM: 32 x 128 bf16 tile, global stride 512
    if (w == 0) {
      tdm_load_2d_bf16((uint32_t)(uintptr_t)Ks,
                       (const unsigned short*)kb + (size_t)kstart * 512 + kvh * 128,
                       128, 32, 512);
    }
    { // V block transposed into Vt[dh][key] (manual: TDM cannot transpose)
      int key = tid >> 3, e = tid & 7;
      const unsigned short* s =
          (const unsigned short*)vb + (size_t)(kstart + key) * 512 + kvh * 128 + e * 16;
#pragma unroll
      for (int j = 0; j < 16; ++j) VtS[(e * 16 + j) * 40 + key] = s[j];
    }
    __builtin_amdgcn_s_wait_tensorcnt(0);  // TDM tiles (Q first iter, K) landed
    __syncthreads();                       // publish to all waves

    // S = Q(16x128) @ K^T -> 16 x 32 per wave
    v8f sa0 = (v8f){0,0,0,0,0,0,0,0};
    v8f sa1 = (v8f){0,0,0,0,0,0,0,0};
#pragma unroll
    for (int kk = 0; kk < 4; ++kk) {
      v16bf aq  = ldsA(Qs + (w * 16 + ll) * 68 + kk * 16, lh);
      v16bf bk0 = ldsB(Ks + (ll) * 68 + kk * 16, lh);
      v16bf bk1 = ldsB(Ks + (16 + ll) * 68 + kk * 16, lh);
      sa0 = wmma_bf16(aq, bk0, sa0);
      sa1 = wmma_bf16(aq, bk1, sa1);
    }

    // scale + window mask + online softmax update
    const int qrow0 = qs + w * 16 + 8 * lh;
    const int kp0 = kstart + ll, kp1 = kstart + 16 + ll;
#pragma unroll
    for (int r = 0; r < 8; ++r) {
      const int qrow = qrow0 + r;
      float s0 = sa0[r] * SCALE, s1 = sa1[r] * SCALE;
      if (kp0 < lo || kp0 > qrow) s0 = -3.0e38f;
      if (kp1 < lo || kp1 > qrow) s1 = -3.0e38f;
      float v = fmaxf(s0, s1);
      v = fmaxf(v, __shfl_xor(v, 1)); v = fmaxf(v, __shfl_xor(v, 2));
      v = fmaxf(v, __shfl_xor(v, 4)); v = fmaxf(v, __shfl_xor(v, 8));
      const float mn = fmaxf(m8[r], v);
      const float alpha = __expf(m8[r] - mn);
      m8[r] = mn;
      const float p0 = __expf(s0 - mn), p1 = __expf(s1 - mn);
      float rs = p0 + p1;
      rs += __shfl_xor(rs, 1); rs += __shfl_xor(rs, 2);
      rs += __shfl_xor(rs, 4); rs += __shfl_xor(rs, 8);
      l8[r] = l8[r] * alpha + rs;
#pragma unroll
      for (int t8 = 0; t8 < 8; ++t8) o[t8][r] *= alpha;
      // transpose P through per-wave LDS (C layout -> A layout)
      const int prow = w * 16 + r + 8 * lh;
      PlsS[prow * 36 + ll]      = f2bf(p0);
      PlsS[prow * 36 + 16 + ll] = f2bf(p1);
    }
    __syncthreads();  // order P writes before fragment reads (and Vt reads)

    // O += P(16x32) @ V(32x128)
    v16bf pA = ldsA(Pls + (w * 16 + ll) * 18, lh);
#pragma unroll
    for (int t8 = 0; t8 < 8; ++t8) {
      v16bf vB = ldsB(Vt + (t8 * 16 + ll) * 20, lh);
      o[t8] = wmma_bf16(pA, vB, o[t8]);
    }
  }

  // epilogue: y[t][h*128+dh] = O / l
  float rl[8];
#pragma unroll
  for (int r = 0; r < 8; ++r) rl[r] = 1.0f / l8[r];
#pragma unroll
  for (int t8 = 0; t8 < 8; ++t8) {
    float* yp = y + (size_t)(qs + w * 16 + 8 * lh) * 2048 + h * 128 + t8 * 16 + ll;
#pragma unroll
    for (int r = 0; r < 8; ++r) yp[(size_t)r * 2048] = o[t8][r] * rl[r];
  }
}

// ---------------------------------------------------------------------------
// launch
// ---------------------------------------------------------------------------
extern "C" void kernel_launch(void* const* d_in, const int* in_sizes, int n_in,
                              void* d_out, int out_size, void* d_ws, size_t ws_size,
                              hipStream_t stream) {
  (void)in_sizes; (void)n_in; (void)out_size; (void)ws_size;
  const float* x  = (const float*)d_in[0];
  const float* wq = (const float*)d_in[1];
  const float* wk = (const float*)d_in[2];
  const float* wv = (const float*)d_in[3];
  const float* wo = (const float*)d_in[4];
  float* out = (float*)d_out;

  char* ws = (char*)d_ws;
  float*    qf  = (float*)(ws + 0);           // 4096*2048*4 = 33554432
  float*    kf  = (float*)(ws + 33554432);    // 4096* 512*4 =  8388608
  float*    vf  = (float*)(ws + 41943040);    // 4096* 512*4 =  8388608
  uint32_t* qbf = (uint32_t*)(ws + 50331648); // 4096*2048*2 = 16777216
  uint32_t* kbf = (uint32_t*)(ws + 67108864); // 4096* 512*2 =  4194304
  uint32_t* vbf = (uint32_t*)(ws + 71303168); // 4096* 512*2 =  4194304
  float*    yb  = (float*)(ws + 75497472);    // 4096*2048*4 = 33554432
  // total workspace: 109,051,904 bytes

  dim3 blk(256);
  // q/k/v projections
  gemm_xwT_bf16<<<dim3(2048 / 64, 4096 / 128), blk, 0, stream>>>(x, wq, qf, 4096, 2048, 2048);
  gemm_xwT_bf16<<<dim3( 512 / 64, 4096 / 128), blk, 0, stream>>>(x, wk, kf, 4096,  512, 2048);
  gemm_xwT_bf16<<<dim3( 512 / 64, 4096 / 128), blk, 0, stream>>>(x, wv, vf, 4096,  512, 2048);
  // rmsnorm+rope (q,k) -> bf16 ; v -> bf16
  rmsnorm_rope_kernel<<<10240, blk, 0, stream>>>(qf, kf, qbf, kbf);
  cvt_f32_bf16x2<<<4096, blk, 0, stream>>>(vf, vbf, 1048576);
  // windowed attention (TDM-fed)
  attn_kernel<<<512, blk, 0, stream>>>(qbf, kbf, vbf, yb);
  // output projection
  gemm_xwT_bf16<<<dim3(2048 / 64, 4096 / 128), blk, 0, stream>>>(yb, wo, out, 4096, 2048, 2048);
}
